// LongformerAttentionMethod_23398981829205
// MI455X (gfx1250) — matrix operand order repaired and checked by reading, not compiled, not run on hardware
//
#include <hip/hip_runtime.h>

// ---------------- problem constants (fixed by setup_inputs) ----------------
#define BATCH   1
#define HEADS   16
#define BHD     (BATCH*HEADS)     // 16
#define SLEN    4096
#define DHEAD   64
#define WIN     256
#define NGLOB   64
#define KB      64                // keys per block iteration
#define QW      16                // query rows per wave (WMMA M)
#define NWAVE   4
#define QB      (QW*NWAVE)        // 64 query rows per workgroup
#define LDK     72                // LDS pitch in halfs (144 B -> 16B aligned rows)
#define NEGINF  (-1.0e30f)

typedef __attribute__((ext_vector_type(16))) _Float16 v16h;
typedef __attribute__((ext_vector_type(8)))  _Float16 v8h;
typedef __attribute__((ext_vector_type(8)))  float    v8f;

// ---------------------------------------------------------------------------
__device__ __forceinline__ v8f wmma16(v16h a, v16h b, v8f c) {
  // (neg_a, A, neg_b, B, c_mod, C, reuse_a, reuse_b)
  return __builtin_amdgcn_wmma_f32_16x16x32_f16(false, a, false, b, (short)0,
                                                c, false, false);
}

// 16-bit A-matrix fragment (16x32): lanes<16 hold K={8hi..8hi+7, 16+8hi..};
// elements j=0..7 -> K = kc*32 + 8*hi + j ; j=8..15 -> K = kc*32 + 16 + 8*hi + (j-8)
__device__ __forceinline__ v16h load_afrag(const _Float16* rowbase, int hi, int kc) {
  const _Float16* p = rowbase + kc * 32 + hi * 8;
  v8h a = *(const v8h*)(p);
  v8h b = *(const v8h*)(p + 16);
  v16h r;
#pragma unroll
  for (int i = 0; i < 8; ++i) { r[i] = a[i]; r[i + 8] = b[i]; }
  return r;
}

// 16-bit B-matrix fragment (32x16): lane n holds col n; elements j=0..7 ->
// K = 8*hi + j, j=8..15 -> K = 16 + 8*hi + (j-8) (within the kc*32 chunk).
__device__ __forceinline__ v16h load_bfrag(const _Float16* colbase, int hi, int kc) {
  const _Float16* p = colbase + kc * 32 + hi * 8;
  v8h a = *(const v8h*)(p);
  v8h b = *(const v8h*)(p + 16);
  v16h r;
#pragma unroll
  for (int i = 0; i < 8; ++i) { r[i] = a[i]; r[i + 8] = b[i]; }
  return r;
}

// Q fragment straight from global fp32 row, scaled by 1/sqrt(D)=0.125, fp16.
__device__ __forceinline__ v16h load_qfrag(const float* __restrict__ qrow,
                                           int hi, int kc) {
  const float scale = 0.125f;
  const int b0 = kc * 32 + hi * 8;
  v16h r;
#pragma unroll
  for (int j = 0; j < 8; ++j) {
    r[j]     = (_Float16)(qrow[b0 + j] * scale);
    r[j + 8] = (_Float16)(qrow[b0 + 16 + j] * scale);
  }
  return r;
}

__device__ __forceinline__ float halfmax16(float v) {
#pragma unroll
  for (int m = 1; m < 16; m <<= 1) v = fmaxf(v, __shfl_xor(v, m, 32));
  return v;
}
__device__ __forceinline__ float halfsum16(float v) {
#pragma unroll
  for (int m = 1; m < 16; m <<= 1) v += __shfl_xor(v, m, 32);
  return v;
}

// Cooperative load of one 64-key tile: K key-major, V dim-major, per-col valid.
__device__ __forceinline__ void load_kv_tile(
    const float* __restrict__ kbase, const float* __restrict__ vbase,
    const int* __restrict__ gmask, const int* __restrict__ pmask,
    const int* __restrict__ gpos, bool globalBlk, bool exclGlob, int kb0,
    _Float16* __restrict__ sK, _Float16* __restrict__ sV,
    int* __restrict__ sOK, int tid) {
  const int krow = tid >> 1;            // 0..63
  const int dimb = (tid & 1) * 32;      // 0 or 32
  int t; bool ok;
  if (globalBlk) {
    t = gpos[krow];
    ok = (pmask[t] != 0);
  } else {
    t = kb0 + krow;
    ok = (t >= 0) && (t < SLEN);
    if (ok) ok = (pmask[t] != 0) && (!exclGlob || gmask[t] == 0);
  }
  const int ts = t < 0 ? 0 : (t >= SLEN ? SLEN - 1 : t);
  const float* kp = kbase + (size_t)ts * DHEAD + dimb;
  const float* vp = vbase + (size_t)ts * DHEAD + dimb;
#pragma unroll
  for (int j = 0; j < 32; ++j) {
    sK[krow * LDK + dimb + j]   = (_Float16)kp[j];
    sV[(dimb + j) * LDK + krow] = (_Float16)vp[j];  // transpose to dim-major
  }
  if ((tid & 1) == 0) sOK[krow] = ok ? 1 : 0;
  if (!globalBlk && ts + KB < SLEN) {               // gfx1250 global_prefetch_b8
    __builtin_prefetch(kp + KB * DHEAD, 0, 1);
    __builtin_prefetch(vp + KB * DHEAD, 0, 1);
  }
}

// One flash-attention step over a 64-key tile: 8 QK WMMAs + 8 PV WMMAs.
// fullBand (wave-uniform): the whole 64-key block is inside the band for
// every row of this wave, so the per-element band test can be skipped.
__device__ __forceinline__ void flash_step(
    v16h qf0, v16h qf1, const _Float16* __restrict__ sK,
    const _Float16* __restrict__ sV, _Float16* __restrict__ sPw,
    const int* __restrict__ sOK, bool band, bool fullBand, int kb0, int qrow0,
    int hi, int ln, v8f acc[4], float* mrun, float* lrun) {
  v8f sc[4];
#pragma unroll
  for (int nt = 0; nt < 4; ++nt) {
    const _Float16* kcol = sK + (nt * 16 + ln) * LDK;
    v8f c = {0.f, 0.f, 0.f, 0.f, 0.f, 0.f, 0.f, 0.f};
    c = wmma16(qf0, load_bfrag(kcol, hi, 0), c);
    c = wmma16(qf1, load_bfrag(kcol, hi, 1), c);
    sc[nt] = c;
  }

  // Wave-uniform fast path: all 64 columns valid AND no band clipping needed.
  const bool ok4 = (sOK[ln] != 0) & (sOK[16 + ln] != 0) &
                   (sOK[32 + ln] != 0) & (sOK[48 + ln] != 0);
  const bool needMask = (!__all(ok4)) || (band && !fullBand);
  if (needMask) {
    // mask in C layout: row = r + 8*hi, col = nt*16 + ln
#pragma unroll
    for (int nt = 0; nt < 4; ++nt) {
      const int col = nt * 16 + ln;
      const bool ok = sOK[col] != 0;
      const int t = kb0 + col;
#pragma unroll
      for (int r = 0; r < 8; ++r) {
        const int srow = qrow0 + r + 8 * hi;
        const bool inband = !band || ((t - srow) <= WIN && (srow - t) <= WIN);
        if (!(ok && inband)) sc[nt][r] = NEGINF;
      }
    }
  }

  // online softmax update + write P (f16) to per-wave LDS pane
#pragma unroll
  for (int r = 0; r < 8; ++r) {
    float bm = fmaxf(fmaxf(sc[0][r], sc[1][r]), fmaxf(sc[2][r], sc[3][r]));
    bm = halfmax16(bm);
    const float nm = fmaxf(mrun[r], bm);
    const float alpha = __expf(mrun[r] - nm);   // finite: NEGINF is -1e30
    mrun[r] = nm;
    lrun[r] *= alpha;
#pragma unroll
    for (int nt = 0; nt < 4; ++nt) acc[nt][r] *= alpha;
    float rs = 0.f;
#pragma unroll
    for (int nt = 0; nt < 4; ++nt) {
      const float sv = sc[nt][r];
      const float p = (sv <= 0.5f * NEGINF) ? 0.f : __expf(sv - nm);
      rs += p;
      sPw[(r + 8 * hi) * LDK + nt * 16 + ln] = (_Float16)p;
    }
    lrun[r] += halfsum16(rs);
  }

  // PV: acc += P(16x64) * V(64x64)   (per-wave LDS pane, same-wave DS ordering)
  const _Float16* prow = sPw + ln * LDK;
#pragma unroll
  for (int kc = 0; kc < 2; ++kc) {
    const v16h pa = load_afrag(prow, hi, kc);
#pragma unroll
    for (int nt = 0; nt < 4; ++nt)
      acc[nt] = wmma16(pa, load_bfrag(sV + (nt * 16 + ln) * LDK, hi, kc), acc[nt]);
  }
}

// ---------------------------------------------------------------------------
__global__ void build_gpos(const int* __restrict__ gmask, int* __restrict__ gpos) {
  if (threadIdx.x == 0 && blockIdx.x == 0) {
    int cnt = 0;
    for (int s = 0; s < SLEN && cnt < NGLOB; ++s)
      if (gmask[s]) gpos[cnt++] = s;
    for (int s = 0; s < SLEN && cnt < NGLOB; ++s)
      if (!gmask[s]) gpos[cnt++] = s;
  }
}

// Kernel A: band + global-key attention for all rows.
__global__ __launch_bounds__(128) void longformer_band(
    const float* __restrict__ q, const float* __restrict__ k,
    const float* __restrict__ v, const int* __restrict__ gmask,
    const int* __restrict__ pmask, const int* __restrict__ gpos,
    float* __restrict__ out) {
  __shared__ __align__(16) _Float16 sK[KB * LDK];
  __shared__ __align__(16) _Float16 sV[DHEAD * LDK];
  __shared__ __align__(16) _Float16 sP[NWAVE * QW * LDK];
  __shared__ int sOK[KB];

  const int bh = blockIdx.y;
  const int q0 = blockIdx.x * QB;
  const int tid = threadIdx.x;
  const int wave = tid >> 5, lane = tid & 31;
  const int hi = lane >> 4, ln = lane & 15;
  const int qrow0 = q0 + wave * QW;

  const float* qbase = q + (size_t)bh * SLEN * DHEAD;
  const float* kbase = k + (size_t)bh * SLEN * DHEAD;
  const float* vbase = v + (size_t)bh * SLEN * DHEAD;
  _Float16* sPw = sP + wave * QW * LDK;

  const float* qrow = qbase + (size_t)(qrow0 + ln) * DHEAD;
  const v16h qf0 = load_qfrag(qrow, hi, 0);
  const v16h qf1 = load_qfrag(qrow, hi, 1);

  const v8f z8 = {0.f, 0.f, 0.f, 0.f, 0.f, 0.f, 0.f, 0.f};
  v8f acc[4] = {z8, z8, z8, z8};
  float mrun[8], lrun[8];
#pragma unroll
  for (int r = 0; r < 8; ++r) { mrun[r] = NEGINF; lrun[r] = 0.f; }

  for (int it = 0; it < 10; ++it) {
    const bool gblk = (it == 0);
    const int kb0 = q0 - WIN + (it - 1) * KB;
    __syncthreads();
    load_kv_tile(kbase, vbase, gmask, pmask, gpos, gblk, /*exclGlob=*/true,
                 kb0, sK, sV, sOK, tid);
    __syncthreads();
    // block cols [kb0, kb0+63] vs wave rows [qrow0, qrow0+15]:
    // fully in-band iff kb0-qrow0 in [15-WIN, WIN-63]  (wave-uniform)
    const int d = kb0 - qrow0;
    const bool fullBand = (d >= 15 - WIN) && (d <= WIN - (KB - 1));
    flash_step(qf0, qf1, sK, sV, sPw, sOK, /*band=*/!gblk, fullBand, kb0,
               qrow0, hi, ln, acc, mrun, lrun);
  }

  float* obase = out + (size_t)bh * SLEN * DHEAD;
#pragma unroll
  for (int r = 0; r < 8; ++r) {
    const int srow = qrow0 + r + 8 * hi;
    const float inv = lrun[r] > 0.f ? 1.f / lrun[r] : 0.f;
#pragma unroll
    for (int nt = 0; nt < 4; ++nt)
      obase[(size_t)srow * DHEAD + nt * 16 + ln] = acc[nt][r] * inv;
  }
}

// Kernel B: full attention for the G global query rows (overwrites kernel A).
__global__ __launch_bounds__(128) void longformer_global_rows(
    const float* __restrict__ q, const float* __restrict__ k,
    const float* __restrict__ v, const int* __restrict__ gmask,
    const int* __restrict__ pmask, const int* __restrict__ gpos,
    float* __restrict__ out) {
  __shared__ __align__(16) _Float16 sK[KB * LDK];
  __shared__ __align__(16) _Float16 sV[DHEAD * LDK];
  __shared__ __align__(16) _Float16 sP[NWAVE * QW * LDK];
  __shared__ int sOK[KB];

  const int bh = blockIdx.x;
  const int tid = threadIdx.x;
  const int wave = tid >> 5, lane = tid & 31;
  const int hi = lane >> 4, ln = lane & 15;

  const float* qbase = q + (size_t)bh * SLEN * DHEAD;
  const float* kbase = k + (size_t)bh * SLEN * DHEAD;
  const float* vbase = v + (size_t)bh * SLEN * DHEAD;
  _Float16* sPw = sP + wave * QW * LDK;

  const int qidx = gpos[wave * QW + ln];           // gathered global row
  const float* qrow = qbase + (size_t)qidx * DHEAD;
  const v16h qf0 = load_qfrag(qrow, hi, 0);
  const v16h qf1 = load_qfrag(qrow, hi, 1);

  const v8f z8 = {0.f, 0.f, 0.f, 0.f, 0.f, 0.f, 0.f, 0.f};
  v8f acc[4] = {z8, z8, z8, z8};
  float mrun[8], lrun[8];
#pragma unroll
  for (int r = 0; r < 8; ++r) { mrun[r] = NEGINF; lrun[r] = 0.f; }

  for (int kb0 = 0; kb0 < SLEN; kb0 += KB) {
    __syncthreads();
    load_kv_tile(kbase, vbase, gmask, pmask, gpos, /*globalBlk=*/false,
                 /*exclGlob=*/false, kb0, sK, sV, sOK, tid);
    __syncthreads();
    flash_step(qf0, qf1, sK, sV, sPw, sOK, /*band=*/false, /*fullBand=*/true,
               kb0, 0, hi, ln, acc, mrun, lrun);
  }

  float* obase = out + (size_t)bh * SLEN * DHEAD;
#pragma unroll
  for (int r = 0; r < 8; ++r) {
    const int orow = gpos[wave * QW + r + 8 * hi];
    const float inv = lrun[r] > 0.f ? 1.f / lrun[r] : 0.f;
#pragma unroll
    for (int nt = 0; nt < 4; ++nt)
      obase[(size_t)orow * DHEAD + nt * 16 + ln] = acc[nt][r] * inv;
  }
}

// ---------------------------------------------------------------------------
extern "C" void kernel_launch(void* const* d_in, const int* in_sizes, int n_in,
                              void* d_out, int out_size, void* d_ws,
                              size_t ws_size, hipStream_t stream) {
  (void)in_sizes; (void)n_in; (void)out_size; (void)ws_size;
  const float* q = (const float*)d_in[0];
  const float* k = (const float*)d_in[1];
  const float* v = (const float*)d_in[2];
  const int* gmask = (const int*)d_in[3];
  const int* pmask = (const int*)d_in[4];
  // d_in[5] = num_global (== NGLOB, compile-time)
  float* out = (float*)d_out;
  int* gpos = (int*)d_ws;                         // NGLOB ints of scratch

  build_gpos<<<1, 32, 0, stream>>>(gmask, gpos);
  dim3 gridA(SLEN / QB, BHD);
  longformer_band<<<gridA, 128, 0, stream>>>(q, k, v, gmask, pmask, gpos, out);
  longformer_global_rows<<<dim3(BHD), 128, 0, stream>>>(q, k, v, gmask, pmask,
                                                        gpos, out);
}
